// AttnLSTMDecoder_18975165513918
// MI455X (gfx1250) — matrix operand order repaired and checked by reading, not compiled, not run on hardware
//
#include <hip/hip_runtime.h>
#include <hip/hip_bf16.h>
#include <math.h>

// ---------- types ----------
typedef __bf16 bf16;
typedef __attribute__((ext_vector_type(16))) __bf16          v16bf;
typedef __attribute__((ext_vector_type(8)))  float           v8f;
typedef __attribute__((ext_vector_type(8)))  unsigned short  u16x8;  // 16B
typedef __attribute__((ext_vector_type(4)))  unsigned short  u16x4;  // 8B

#define DEV static __device__ __forceinline__

DEV unsigned short f2bf_bits(float f) {
  unsigned int u = __builtin_bit_cast(unsigned int, f);
  u += 0x7FFFu + ((u >> 16) & 1u);            // round-to-nearest-even
  return (unsigned short)(u >> 16);
}
DEV bf16  f2bf(float f) { return __builtin_bit_cast(bf16, f2bf_bits(f)); }
DEV float bf2f(unsigned short b) {
  unsigned int u = ((unsigned int)b) << 16;
  return __builtin_bit_cast(float, u);
}
DEV float sigmoidf_(float x) { return 1.0f / (1.0f + expf(-x)); }

// ---------------------------------------------------------------------------
// Generic bf16 WMMA GEMM:  C[M,N] = act( A[M,K] * W[N,K]^T + bias )
// A, W row-major bf16. Block = 256 threads = 8 waves as 2(M) x 4(N).
// Wave tile = 16(M) x 64(N), K step = 32 (v_wmma_f32_16x16x32_bf16).
//
// Operand layouts per CDNA5 ISA 7.12.2 (wave32):
//  A 16x32: lane&15 = row M; lanes<16 hold K = kb..kb+7 and kb+16..kb+23
//           (kb = (lane>>4)*8); two contiguous 16B loads at +0 and +16 halves.
//  B 32x16 (col-major from row-major W): lane&15 = col N; same K split as A
//           (kb..kb+7 and kb+16..kb+23); two contiguous 16B loads.
//  D f32 16x16: n = lane&15 (+0/+16 group), m = r + 8*(lane>>4), r = VGPR idx.
// ---------------------------------------------------------------------------
template <int OUT_BF16, int HAS_BIAS, int DO_TANH>
__global__ __launch_bounds__(256)
void wmma_gemm(const bf16* __restrict__ A, const bf16* __restrict__ W,
               float* __restrict__ Cf, bf16* __restrict__ Cb,
               const float* __restrict__ bias,
               int M, int N, int K, int lda, int ldw, int ldc) {
  const int lane = threadIdx.x & 31;
  const int wave = threadIdx.x >> 5;
  const int m0 = blockIdx.y * 32 + (wave & 1) * 16;
  const int n0 = blockIdx.x * 256 + (wave >> 1) * 64;
  if (m0 >= M || n0 >= N) return;        // wave-uniform: EXEC stays all-ones

  v8f acc[4] = {v8f{}, v8f{}, v8f{}, v8f{}};

  const int arow = m0 + (lane & 15);
  const int akb  = (lane >> 4) << 3;     // 0 or 8
  const int bkb  = (lane >> 4) << 3;     // 0 or 8
  const long ldaL = lda, ldwL = ldw;

  for (int k0 = 0; k0 < K; k0 += 32) {
    const bf16* pa = A + (long)arow * ldaL + (k0 + akb);
    union { v16bf v; u16x8 s[2]; } ua;
    ua.s[0] = *(const u16x8*)(pa);
    ua.s[1] = *(const u16x8*)(pa + 16);
    __builtin_prefetch(pa + 64, 0, 1);   // gfx1250 global_prefetch_b8, K tile two steps ahead
#pragma unroll
    for (int j = 0; j < 4; ++j) {
      const bf16* pb = W + (long)(n0 + j * 16 + (lane & 15)) * ldwL + (k0 + bkb);
      union { v16bf v; u16x8 s[2]; } ub;
      ub.s[0] = *(const u16x8*)(pb);
      ub.s[1] = *(const u16x8*)(pb + 16);
      acc[j] = __builtin_amdgcn_wmma_f32_16x16x32_bf16(
          /*neg_a=*/false, ua.v, /*neg_b=*/false, ub.v,
          /*c_mod=*/(short)0, acc[j], /*reuse_a=*/false, /*reuse_b=*/false);
    }
  }

  const int mbase = m0 + ((lane >> 4) << 3);
#pragma unroll
  for (int j = 0; j < 4; ++j) {
    const int n = n0 + j * 16 + (lane & 15);
    float bv = HAS_BIAS ? bias[n] : 0.0f;
#pragma unroll
    for (int r = 0; r < 8; ++r) {
      const int m = mbase + r;
      if (m < M && n < N) {
        float v = acc[j][r] + bv;
        if (DO_TANH) v = tanhf(v);
        if (OUT_BF16) Cb[(long)m * ldc + n] = f2bf(v);
        else          Cf[(long)m * ldc + n] = v;
      }
    }
  }
}

// ---------- elementwise / pack kernels ----------
__global__ void cvt_kernel(bf16* __restrict__ dst, const float* __restrict__ src, long n) {
  long i = (long)blockIdx.x * blockDim.x + threadIdx.x;
  if (i < n) dst[i] = f2bf(src[i]);
}

__global__ void copyf_kernel(float* __restrict__ dst, const float* __restrict__ src, long n) {
  long i = (long)blockIdx.x * blockDim.x + threadIdx.x;
  if (i < n) dst[i] = src[i];
}

// Wcat[n, 0:Kh] = W_ih[n, :], Wcat[n, Kh:2Kh] = W_hh[n, :]   (Kh = 1024)
__global__ void wcat_kernel(bf16* __restrict__ Wc, const float* __restrict__ Wih,
                            const float* __restrict__ Whh, long N, int Kh) {
  long i = (long)blockIdx.x * blockDim.x + threadIdx.x;
  long tot = N * (long)(2 * Kh);
  if (i >= tot) return;
  long n = i / (2 * Kh);
  int  k = (int)(i % (2 * Kh));
  float v = (k < Kh) ? Wih[n * Kh + k] : Whh[n * Kh + (k - Kh)];
  Wc[i] = f2bf(v);
}

__global__ void bias_kernel(float* bs, const float* b1, const float* b2, int n) {
  int i = blockIdx.x * blockDim.x + threadIdx.x;
  if (i < n) bs[i] = b1[i] + b2[i];
}

// xcat[b, 0:512]=target[b,t,:], [512:1024]=prev_out, [1024:2048]=h  (row = 2048)
__global__ void pack_xcat_kernel(bf16* __restrict__ xcat, const float* __restrict__ target,
                                 const float* __restrict__ prev, const float* __restrict__ h,
                                 int t, int T, int D, int H) {
  int i = blockIdx.x * blockDim.x + threadIdx.x;  // B*2048
  int b = i >> 11, j = i & 2047;
  float v;
  if (j < D)           v = target[((long)b * T + t) * D + j];
  else if (j < 2 * D)  v = prev[(long)b * D + (j - D)];
  else                 v = h[(long)b * H + (j - 2 * D)];
  xcat[i] = f2bf(v);
}

// dst[b, 0:H]=a[b,:], dst[b, H:2H]=c[b,:]   (row = 2048, H = 1024)
__global__ void pack2_kernel(bf16* __restrict__ dst, const float* __restrict__ a,
                             const float* __restrict__ c, int H) {
  int i = blockIdx.x * blockDim.x + threadIdx.x;  // B*2048
  int b = i >> 11, j = i & 2047;
  float v = (j < H) ? a[(long)b * H + j] : c[(long)b * H + (j - H)];
  dst[i] = f2bf(v);
}

// torch gate order i,f,g,o as contiguous H-chunks of the 4H gate vector
__global__ void lstm_cell_kernel(const float* __restrict__ gates, float* __restrict__ h,
                                 float* __restrict__ c, bf16* __restrict__ h2b, int H) {
  int idx = blockIdx.x * blockDim.x + threadIdx.x;  // B*H
  int b = idx / H, n = idx % H;
  const float* g = gates + (long)b * 4 * H;
  float c2 = sigmoidf_(g[n + H]) * c[idx] + sigmoidf_(g[n]) * tanhf(g[n + 2 * H]);
  float h2 = sigmoidf_(g[n + 3 * H]) * tanhf(c2);
  c[idx] = c2;
  h[idx] = h2;
  h2b[idx] = f2bf(h2);
}

// scores[b,s] = dot(h2[b,:], proj[b,s,:]) — one wave32 per (b,s)
__global__ __launch_bounds__(256)
void scores_kernel(const bf16* __restrict__ proj, const bf16* __restrict__ h2,
                   float* __restrict__ scores, int S, int H) {
  int gw = blockIdx.x * 8 + (threadIdx.x >> 5);
  int lane = threadIdx.x & 31;
  int b = gw / S, s = gw % S;
  const bf16* pp = proj + ((long)b * S + s) * H;
  const bf16* ph = h2 + (long)b * H;
  float acc = 0.0f;
  for (int h0 = lane * 4; h0 < H; h0 += 128) {
    u16x4 a = *(const u16x4*)(pp + h0);
    u16x4 x = *(const u16x4*)(ph + h0);
#pragma unroll
    for (int e = 0; e < 4; ++e) acc += bf2f(a[e]) * bf2f(x[e]);
  }
#pragma unroll
  for (int off = 16; off > 0; off >>= 1) acc += __shfl_xor(acc, off, 32);
  if (lane == 0) scores[(long)b * S + s] = acc;
}

// masked softmax over S; mask rebuilt from source_length (s >= len -> weight 0)
__global__ __launch_bounds__(256)
void softmax_kernel(const float* __restrict__ scores, const int* __restrict__ slen,
                    float* __restrict__ attw, int S) {
  __shared__ float red[256];
  int b = blockIdx.x, tid = threadIdx.x;
  int len = slen[b];
  const float* sc = scores + (long)b * S;
  float mx = -3.4e38f;
  for (int s = tid; s < S; s += 256) if (s < len) mx = fmaxf(mx, sc[s]);
  red[tid] = mx; __syncthreads();
  for (int o = 128; o > 0; o >>= 1) { if (tid < o) red[tid] = fmaxf(red[tid], red[tid + o]); __syncthreads(); }
  mx = red[0]; __syncthreads();
  float sum = 0.0f;
  for (int s = tid; s < S; s += 256) if (s < len) sum += expf(sc[s] - mx);
  red[tid] = sum; __syncthreads();
  for (int o = 128; o > 0; o >>= 1) { if (tid < o) red[tid] += red[tid + o]; __syncthreads(); }
  float inv = 1.0f / red[0];
  float* aw = attw + (long)b * S;
  for (int s = tid; s < S; s += 256) aw[s] = (s < len) ? expf(sc[s] - mx) * inv : 0.0f;
}

// ctx[b,h] = sum_s attw[b,s] * enc[b,s,h]; attention weights staged in LDS
__global__ __launch_bounds__(256)
void context_kernel(const bf16* __restrict__ enc, const float* __restrict__ attw,
                    float* __restrict__ ctx, int S, int H) {
  __shared__ float aw[1024];
  int b = blockIdx.x, tid = threadIdx.x;
  for (int s = tid; s < S; s += 256) aw[s] = attw[(long)b * S + s];
  __syncthreads();
  for (int h = tid; h < H; h += 256) {
    float acc = 0.0f;
    const bf16* pe = enc + (long)b * S * H + h;
    for (int s = 0; s < S; ++s) acc += aw[s] * bf2f(pe[(long)s * H]);
    ctx[(long)b * H + h] = acc;
  }
}

// smean[b,h] = sum_{s<len} enc[b,s,h] / len  (fp32 source for accuracy)
__global__ __launch_bounds__(256)
void seqmean_kernel(const float* __restrict__ enc, const int* __restrict__ slen,
                    float* __restrict__ smean, int S, int H) {
  int b = blockIdx.x, tid = threadIdx.x;
  int len = slen[b];
  float inv = 1.0f / (float)len;
  for (int h = tid; h < H; h += 256) {
    float acc = 0.0f;
    const float* pe = enc + (long)b * S * H + h;
    for (int s = 0; s < len; ++s) acc += pe[(long)s * H];
    smean[(long)b * H + h] = acc * inv;
  }
}

__global__ void scatter_out_kernel(float* __restrict__ out, const float* __restrict__ prev,
                                   int t, int T, int D) {
  int i = blockIdx.x * blockDim.x + threadIdx.x;  // B*D
  int b = i / D, d = i % D;
  out[((long)b * T + t) * D + d] = prev[i];
}

// ---------------------------------------------------------------------------
extern "C" void kernel_launch(void* const* d_in, const int* in_sizes, int n_in,
                              void* d_out, int out_size, void* d_ws, size_t ws_size,
                              hipStream_t stream) {
  (void)in_sizes; (void)n_in; (void)out_size;
  const int B = 64, S = 1024, T = 64, H = 1024, D = 512;

  const float* target = (const float*)d_in[0];
  const float* enc    = (const float*)d_in[1];
  const float* init_h = (const float*)d_in[2];   // [1,B,H] contiguous
  const float* init_c = (const float*)d_in[3];
  const float* W_ih   = (const float*)d_in[4];
  const float* W_hh   = (const float*)d_in[5];
  const float* b_ih   = (const float*)d_in[6];
  const float* b_hh   = (const float*)d_in[7];
  const float* attn_W = (const float*)d_in[8];
  const float* p1W    = (const float*)d_in[9];
  const float* p1b    = (const float*)d_in[10];
  const float* p2W    = (const float*)d_in[11];
  const int*   slen   = (const int*)d_in[12];    // source_length; bool mask [13] unused
  float* out = (float*)d_out;

  char* ws = (char*)d_ws;
  size_t off = 0;
  auto alloc = [&](size_t bytes) -> void* {
    void* p = ws + off;
    off = (off + bytes + 255) & ~(size_t)255;
    return p;
  };
  bf16*  enc_b  = (bf16*)alloc((size_t)B * S * H * 2);       // 134 MB
  bf16*  proj_b = (bf16*)alloc((size_t)B * S * H * 2);       // 134 MB
  bf16*  Wcat_b = (bf16*)alloc((size_t)4 * H * 2048 * 2);    // 16.8 MB
  bf16*  attn_b = (bf16*)alloc((size_t)H * H * 2);
  bf16*  p1_b   = (bf16*)alloc((size_t)H * 2 * H * 2);
  bf16*  p2_b   = (bf16*)alloc((size_t)D * H * 2);
  float* bias4  = (float*)alloc((size_t)4 * H * 4);
  float* gates  = (float*)alloc((size_t)B * 4 * H * 4);
  bf16*  xcat   = (bf16*)alloc((size_t)B * 2048 * 2);
  bf16*  hcb    = (bf16*)alloc((size_t)B * 2048 * 2);
  bf16*  t1b    = (bf16*)alloc((size_t)B * H * 2);
  float* hst    = (float*)alloc((size_t)B * H * 4);
  float* cst    = (float*)alloc((size_t)B * H * 4);
  bf16*  h2b    = (bf16*)alloc((size_t)B * H * 2);
  float* prev   = (float*)alloc((size_t)B * D * 4);
  float* scoresb= (float*)alloc((size_t)B * S * 4);
  float* attw   = (float*)alloc((size_t)B * S * 4);
  float* ctx    = (float*)alloc((size_t)B * H * 4);
  float* smean  = (float*)alloc((size_t)B * H * 4);
  if (off > ws_size) return;   // workspace too small: do nothing (deterministic)

  // ---- one-time conversions / packing ----
  const long nEnc = (long)B * S * H;
  cvt_kernel<<<(unsigned)((nEnc + 255) / 256), 256, 0, stream>>>(enc_b, enc, nEnc);
  cvt_kernel<<<(unsigned)(((long)H * H + 255) / 256), 256, 0, stream>>>(attn_b, attn_W, (long)H * H);
  cvt_kernel<<<(unsigned)(((long)H * 2 * H + 255) / 256), 256, 0, stream>>>(p1_b, p1W, (long)H * 2 * H);
  cvt_kernel<<<(unsigned)(((long)D * H + 255) / 256), 256, 0, stream>>>(p2_b, p2W, (long)D * H);
  wcat_kernel<<<(unsigned)(((long)4 * H * 2048 + 255) / 256), 256, 0, stream>>>(Wcat_b, W_ih, W_hh, 4 * H, H);
  bias_kernel<<<(4 * H + 255) / 256, 256, 0, stream>>>(bias4, b_ih, b_hh, 4 * H);
  copyf_kernel<<<(B * H + 255) / 256, 256, 0, stream>>>(hst, init_h, (long)B * H);
  copyf_kernel<<<(B * H + 255) / 256, 256, 0, stream>>>(cst, init_c, (long)B * H);

  // ---- proj[b,s,:] = attn_W @ enc[b,s,:]  ==  enc @ attn_W^T (bulk WMMA GEMM) ----
  { dim3 g(H / 256, (B * S) / 32);
    wmma_gemm<1, 0, 0><<<g, 256, 0, stream>>>(enc_b, attn_b, nullptr, proj_b, nullptr,
                                              B * S, H, H, H, H, H); }

  // ---- init_out = out_proj(concat(init_h, seq_mean)) ----
  seqmean_kernel<<<B, 256, 0, stream>>>(enc, slen, smean, S, H);
  pack2_kernel<<<(B * 2048) / 256, 256, 0, stream>>>(hcb, init_h, smean, H);
  { dim3 g(H / 256, B / 32);
    wmma_gemm<1, 1, 1><<<g, 256, 0, stream>>>(hcb, p1_b, nullptr, t1b, p1b,
                                              B, H, 2 * H, 2 * H, 2 * H, H); }
  { dim3 g(D / 256, B / 32);
    wmma_gemm<0, 0, 0><<<g, 256, 0, stream>>>(t1b, p2_b, prev, nullptr, nullptr,
                                              B, D, H, H, H, D); }

  // ---- recurrence: 64 timesteps ----
  for (int t = 0; t < T; ++t) {
    pack_xcat_kernel<<<(B * 2048) / 256, 256, 0, stream>>>(xcat, target, prev, hst, t, T, D, H);
    { dim3 g(4 * H / 256, B / 32);   // gates = xcat @ [W_ih;W_hh]^T + (b_ih+b_hh)
      wmma_gemm<0, 1, 0><<<g, 256, 0, stream>>>(xcat, Wcat_b, gates, nullptr, bias4,
                                                B, 4 * H, 2048, 2048, 2048, 4 * H); }
    lstm_cell_kernel<<<(B * H) / 256, 256, 0, stream>>>(gates, hst, cst, h2b, H);
    scores_kernel<<<(B * S) / 8, 256, 0, stream>>>(proj_b, h2b, scoresb, S, H);
    softmax_kernel<<<B, 256, 0, stream>>>(scoresb, slen, attw, S);
    context_kernel<<<B, 256, 0, stream>>>(enc_b, attw, ctx, S, H);
    pack2_kernel<<<(B * 2048) / 256, 256, 0, stream>>>(hcb, hst, ctx, H);
    { dim3 g(H / 256, B / 32);       // t1 = tanh(hc @ proj1^T + b1)
      wmma_gemm<1, 1, 1><<<g, 256, 0, stream>>>(hcb, p1_b, nullptr, t1b, p1b,
                                                B, H, 2 * H, 2 * H, 2 * H, H); }
    { dim3 g(D / 256, B / 32);       // out_t = t1 @ proj2^T
      wmma_gemm<0, 0, 0><<<g, 256, 0, stream>>>(t1b, p2_b, prev, nullptr, nullptr,
                                                B, D, H, H, H, D); }
    scatter_out_kernel<<<(B * D) / 256, 256, 0, stream>>>(out, prev, t, T, D);
  }
}
